// LevelAttention_loss_8847632630341
// MI455X (gfx1250) — compile-verified, matrix operand order checked by reading, not run on hardware
//
#include <hip/hip_runtime.h>
#include <hip/hip_bf16.h>
#include <math.h>

typedef __attribute__((ext_vector_type(16))) _Float16 v16h;
typedef __attribute__((ext_vector_type(8)))  float    v8f;

#define AHW 256            // attention grid height == width (reference constant)
#define WAVES_PER_BLK 8
#define TPW 8              // 16x16 tiles per wave (one half-column)
#define NSTRIP 2           // (256/16) / TPW

// ---------------------------------------------------------------------------
// Kernel 1: per-box rasterization parameters + bucketing by batch.
// Packs [lo | hi<<16] half-open intervals; invalid boxes -> (0,0) = empty
// (empty MUST be lo==hi so that the unsigned-range test (v-lo) < (hi-lo)
// matches nothing).
// ---------------------------------------------------------------------------
__global__ void prep_kernel(const float* __restrict__ target, int nbox, int B,
                            const int* __restrict__ imgh_p,
                            const int* __restrict__ imgw_p,
                            unsigned* __restrict__ offs,   // [B+1]
                            float*    __restrict__ weight, // [B]
                            unsigned* __restrict__ rawYY,
                            unsigned* __restrict__ rawXX,
                            unsigned* __restrict__ rawB,
                            unsigned* __restrict__ listYY,
                            unsigned* __restrict__ listXX)
{
    __shared__ unsigned cnt[64];
    const int tid = threadIdx.x;
    if (tid < 64) cnt[tid] = 0u;
    __syncthreads();

    const float img_w = (float)*imgw_p;
    const float img_h = (float)*imgh_p;
    const float sx = (float)AHW / img_w;
    const float sy = (float)AHW / img_h;

    for (int i = tid; i < nbox; i += blockDim.x) {
        const float* t = target + i * 6;
        const int   b  = (int)t[0];
        const float xc = t[2], yc = t[3], bw = t[4], bh = t[5];
        const float x1 = img_w * (xc - bw * 0.5f);
        const float y1 = img_h * (yc - bh * 0.5f);
        const float x2 = img_w * (xc + bw * 0.5f);
        const float y2 = img_h * (yc + bh * 0.5f);
        const bool valid = (x1 <= img_w) && (y1 <= img_h) &&
                           (x2 <= img_w) && (y2 <= img_h);
        const int X1 = (int)fmaxf(truncf(x1 * sx), 0.0f);
        const int Y1 = (int)fmaxf(truncf(y1 * sy), 0.0f);
        const int X2 = (int)fminf(ceilf(x2 * sx) + 1.0f, (float)AHW);
        const int Y2 = (int)fminf(ceilf(y2 * sy) + 1.0f, (float)AHW);
        unsigned yy = 0u, xx = 0u;                 // empty interval (lo==hi==0)
        if (valid) {
            yy = (unsigned)Y1 | ((unsigned)Y2 << 16);
            xx = (unsigned)X1 | ((unsigned)X2 << 16);
        }
        rawYY[i] = yy; rawXX[i] = xx; rawB[i] = (unsigned)b;
        atomicAdd(&cnt[b], 1u);
    }
    __syncthreads();

    if (tid == 0) {
        unsigned run = 0;
        for (int b = 0; b < B; ++b) {
            offs[b]   = run;
            weight[b] = cnt[b] ? (1.0f / (float)(AHW * AHW)) : 0.0f;
            run += cnt[b];
        }
        offs[B] = run;
    }
    __syncthreads();
    if (tid < B) cnt[tid] = offs[tid];   // reuse as scatter cursors
    __syncthreads();

    for (int i = tid; i < nbox; i += blockDim.x) {
        const unsigned b   = rawB[i];
        const unsigned pos = atomicAdd(&cnt[b], 1u);
        listYY[pos] = rawYY[i];
        listXX[pos] = rawXX[i];
    }
}

// ---------------------------------------------------------------------------
// Kernel 2: one wave per (batch, tx, half-column) strip of 8 output tiles
// (128 rows x 16 cols) of cover = InY^T * InX via v_wmma_f32_16x16x32_f16,
// fused with BCE against the logits.
//
// A-frag element e (lane group g = lane/16): k = g*8 + (e&7) + ((e&8)<<1).
// Any consistent k permutation between A and B is exact (sum over k).
// The B fragment and the shuffled interval data are built once per chunk and
// reused across all 8 tiles (8 WMMAs per chunk).
// ---------------------------------------------------------------------------
__global__ void __launch_bounds__(256)
cover_bce_kernel(const float*    __restrict__ att,
                 const unsigned* __restrict__ offs,
                 const float*    __restrict__ weight,
                 const unsigned* __restrict__ listYY,
                 const unsigned* __restrict__ listXX,
                 float*          __restrict__ waveSum)
{
    const int lane  = threadIdx.x & 31;
    const int wave  = threadIdx.x >> 5;
    const int wid   = blockIdx.x * WAVES_PER_BLK + wave;
    const int batch = wid / (16 * NSTRIP);
    const int rem   = wid % (16 * NSTRIP);
    const int tx    = rem >> 1;          // NSTRIP == 2
    const int ty0   = (rem & 1) * TPW;
    const int g     = lane >> 4;
    const int mn    = lane & 15;
    const int xn    = tx * 16 + mn;      // B col (x)

    const int kbeg = (int)offs[batch];
    const int kend = (int)offs[batch + 1];

    v8f c[TPW];
#pragma unroll
    for (int t = 0; t < TPW; ++t) c[t] = (v8f){};

    for (int base = kbeg; base < kend; base += 32) {
        const int idx = base + lane;
        unsigned yy = 0u, xx = 0u;       // pad with empty boxes
        if (idx < kend) { yy = listYY[idx]; xx = listXX[idx]; }

        unsigned y1[16], ylen[16];
        v16h b;
#pragma unroll
        for (int e = 0; e < 16; ++e) {
            const int src = (e & 7) + (g << 3) + ((e & 8) << 1);
            const unsigned yyk = (unsigned)__shfl((int)yy, src, 32);
            const unsigned xxk = (unsigned)__shfl((int)xx, src, 32);
            y1[e]   = yyk & 0xffffu;
            ylen[e] = (yyk >> 16) - y1[e];
            const unsigned x1   = xxk & 0xffffu;
            const unsigned xlen = (xxk >> 16) - x1;
            b[e] = ((unsigned)(xn - x1) < xlen) ? (_Float16)1.0f
                                                : (_Float16)0.0f;
        }
#pragma unroll
        for (int t = 0; t < TPW; ++t) {
            const int ym = (ty0 + t) * 16 + mn;
            v16h a;
#pragma unroll
            for (int e = 0; e < 16; ++e)
                a[e] = ((unsigned)(ym - y1[e]) < ylen[e]) ? (_Float16)1.0f
                                                          : (_Float16)0.0f;
            c[t] = __builtin_amdgcn_wmma_f32_16x16x32_f16(
                       /*neg_a=*/false, a, /*neg_b=*/false, b,
                       /*c_mod=*/(short)0, c[t],
                       /*reuse_a=*/false, /*reuse_b=*/false);
        }
    }

    // Fused BCE: tile t, lane L, VGPR v holds cover at
    // (y = (ty0+t)*16 + v + 8*g, x = tx*16 + L%16)
    float s = 0.0f;
#pragma unroll
    for (int t = 0; t < TPW; ++t) {
#pragma unroll
        for (int v = 0; v < 8; ++v) {
            const int   y = (ty0 + t) * 16 + v + 8 * g;
            const float l = att[(batch * AHW + y) * AHW + xn];
            const float m = (c[t][v] > 0.0f) ? 1.0f : 0.0f;
            s += fmaxf(l, 0.0f) - l * m + log1pf(expf(-fabsf(l)));
        }
    }
    s *= weight[batch];

    // deterministic in-wave reduction (fixed xor tree)
#pragma unroll
    for (int off = 16; off > 0; off >>= 1)
        s += __shfl_xor(s, off, 32);
    if (lane == 0) waveSum[wid] = s;
}

// ---------------------------------------------------------------------------
// Kernel 3: deterministic fixed-order final reduction.
// ---------------------------------------------------------------------------
__global__ void reduce_kernel(const float* __restrict__ waveSum, int n,
                              float* __restrict__ out)
{
    __shared__ float sh[256];
    const int tid = threadIdx.x;
    float s = 0.0f;
    for (int i = tid; i < n; i += 256) s += waveSum[i];
    sh[tid] = s;
    __syncthreads();
    for (int off = 128; off > 0; off >>= 1) {
        if (tid < off) sh[tid] += sh[tid + off];
        __syncthreads();
    }
    if (tid == 0) out[0] = sh[0];
}

// ---------------------------------------------------------------------------
extern "C" void kernel_launch(void* const* d_in, const int* in_sizes, int n_in,
                              void* d_out, int out_size, void* d_ws, size_t ws_size,
                              hipStream_t stream) {
    const float* att    = (const float*)d_in[0];
    const float* target = (const float*)d_in[1];
    const int*   imgh_p = (const int*)d_in[3];
    const int*   imgw_p = (const int*)d_in[4];

    const int B    = in_sizes[0] / (AHW * AHW);
    const int nbox = in_sizes[1] / 6;

    // workspace layout
    unsigned* offs    = (unsigned*)d_ws;            // 32 slots (B+1 used)
    float*    weight  = (float*)d_ws + 32;          // 32 slots (B used)
    unsigned* rawYY   = (unsigned*)d_ws + 64;
    unsigned* rawXX   = rawYY + nbox;
    unsigned* rawB    = rawXX + nbox;
    unsigned* listYY  = rawB + nbox;
    unsigned* listXX  = listYY + nbox;
    float*    waveSum = (float*)(listXX + nbox);    // B*16*NSTRIP entries

    prep_kernel<<<1, 256, 0, stream>>>(target, nbox, B, imgh_p, imgw_p,
                                       offs, weight,
                                       rawYY, rawXX, rawB, listYY, listXX);

    const int nwaves = B * 16 * NSTRIP;
    cover_bce_kernel<<<nwaves / WAVES_PER_BLK, 32 * WAVES_PER_BLK, 0, stream>>>(
        att, offs, weight, listYY, listXX, waveSum);

    reduce_kernel<<<1, 256, 0, stream>>>(waveSum, nwaves, (float*)d_out);
}
